// NonLocalBlock_82094004896485
// MI455X (gfx1250) — matrix operand order repaired and checked by reading, not compiled, not run on hardware
//
#include <hip/hip_runtime.h>
#include <stdint.h>

#define Bb 4
#define Cc 256
#define Hh 64
#define Ww 64
#define Nn 4096
#define KBIAS (1.5f / 89.09545442950498f)   // (hi-lo)/dist_max, dist_max = 63*sqrt(2)

typedef __attribute__((ext_vector_type(16))) __bf16 v16bf;
typedef __attribute__((ext_vector_type(8)))  float  v8f;

union BF16x16 { v16bf v; uint32_t u[8]; };
union F32x8   { v8f v; float f[8]; };

// Force all 8 fragments to be simultaneously live in VGPRs at this point so the
// scheduler cannot collapse them into one buffer (load->wait->wmma serialization).
#define FORCE_LIVE8(a) asm volatile("" \
  : "+v"(a[0]), "+v"(a[1]), "+v"(a[2]), "+v"(a[3]), \
    "+v"(a[4]), "+v"(a[5]), "+v"(a[6]), "+v"(a[7]))

__device__ __forceinline__ float fast_sqrtf(float x) {
  return __builtin_amdgcn_sqrtf(x);            // single v_sqrt_f32 (TRANS op)
}

__device__ __forceinline__ v8f wmma_bf16(v16bf a, v16bf b, v8f c) {
  // D = A(16x32 bf16) * B(32x16 bf16) + C(16x16 f32)
  return __builtin_amdgcn_wmma_f32_16x16x32_bf16(false, a, false, b, (short)0, c, false, false);
}

__device__ __forceinline__ uint16_t f2bf(float f) {
  uint32_t u = __float_as_uint(f);
  uint32_t r = u + 0x7FFFu + ((u >> 16) & 1u);   // round-to-nearest-even
  return (uint16_t)(r >> 16);
}

// A-matrix fragment (16 rows in lanes, per CDNA5 16-bit A 16x32 layout):
// lane half h, VGPR j holds K = k0 + 8*h + 16*(j>=4) + 2*(j&3) (+1 in hi half)
__device__ __forceinline__ v16bf load_a_frag(const uint16_t* __restrict__ base, int ld,
                                             int k0, int lane) {
  int half = lane >> 4, m = lane & 15;
  const uint16_t* row = base + (size_t)m * ld + k0 + 8 * half;
  BF16x16 r;
#pragma unroll
  for (int j = 0; j < 8; ++j) {
    int off = 16 * (j >> 2) + 2 * (j & 3);       // merges into 2x b128 loads
    r.u[j] = *(const uint32_t*)(row + off);
  }
  return r.v;
}

// B-matrix fragment (KxN, col = lane&15): lane holds K = k0+8*half+0..7 and k0+16+8*half+0..7.
// Source layout is "Bt": [col][K] row-major with leading dim ld.
__device__ __forceinline__ v16bf load_b_frag(const uint16_t* __restrict__ base, int ld,
                                             int k0, int lane) {
  int half = lane >> 4, n = lane & 15;
  const uint16_t* p = base + (size_t)n * ld + k0 + 8 * half;
  BF16x16 r;
  uint4 lo = *(const uint4*)(p);
  uint4 hi = *(const uint4*)(p + 16);
  r.u[0] = lo.x; r.u[1] = lo.y; r.u[2] = lo.z; r.u[3] = lo.w;
  r.u[4] = hi.x; r.u[5] = hi.y; r.u[6] = hi.z; r.u[7] = hi.w;
  return r.v;
}

// ---------------- Kernel 1: fp32 projections, output bf16 -------------------
// theta[n][d], phiT[m][d] stored [N][C]; g stored [C][N].
__global__ __launch_bounds__(256) void proj_kernel(
    const float* __restrict__ x, const float* __restrict__ w_phi,
    const float* __restrict__ w_theta, const float* __restrict__ w_g,
    uint16_t* __restrict__ thetaB, uint16_t* __restrict__ phiB,
    uint16_t* __restrict__ gB) {
  __shared__ float sx[64][68];                 // 68 keeps rows 16B-aligned for b128
  __shared__ float sw[64][68];
  const int n0 = blockIdx.x * 64;
  const int proj = blockIdx.y >> 2;            // 0=theta, 1=phi, 2=g
  const int d0 = (blockIdx.y & 3) * 64;
  const int b = blockIdx.z;
  const float* W = (proj == 0) ? w_theta : ((proj == 1) ? w_phi : w_g);
  const int tid = threadIdx.x;
  const int ti = tid >> 4, tj = tid & 15;      // 16x16 thread tile, 4x4 micro-tile
  const int li = tid >> 4, lj = (tid & 15) * 4;
  float acc[4][4] = {};
  for (int cc = 0; cc < Cc; cc += 64) {
#pragma unroll
    for (int r = 0; r < 4; ++r) {
      int i = li + r * 16;
      *(float4*)&sx[i][lj] = *(const float4*)&x[((size_t)(b * Cc + cc + i)) * Nn + n0 + lj];
      *(float4*)&sw[i][lj] = *(const float4*)&W[(size_t)(d0 + i) * Cc + cc + lj];
    }
    __syncthreads();
#pragma unroll 8
    for (int k = 0; k < 64; ++k) {
      float wv[4], xv[4];
#pragma unroll
      for (int q = 0; q < 4; ++q) { wv[q] = sw[ti * 4 + q][k]; xv[q] = sx[k][tj * 4 + q]; }
#pragma unroll
      for (int q = 0; q < 4; ++q)
#pragma unroll
        for (int r = 0; r < 4; ++r) acc[q][r] += wv[q] * xv[r];
    }
    __syncthreads();
  }
#pragma unroll
  for (int q = 0; q < 4; ++q)
#pragma unroll
    for (int r = 0; r < 4; ++r) {
      int d = d0 + ti * 4 + q, n = n0 + tj * 4 + r;
      uint16_t hv = f2bf(acc[q][r]);
      if (proj == 0)      thetaB[((size_t)b * Nn + n) * Cc + d] = hv;
      else if (proj == 1) phiB[((size_t)b * Nn + n) * Cc + d] = hv;
      else                gB[((size_t)b * Cc + d) * Nn + n] = hv;
    }
}

// Online softmax-stat update for one 16x16 S tile held in acc (D layout).
__device__ __forceinline__ void tile_stats(const F32x8& acc, int n0, int half,
                                           int mi, int mj, float& runmax, float& runsum) {
  float sv[8], tmax = -3.0e38f;
#pragma unroll
  for (int v = 0; v < 8; ++v) {
    int n = n0 + v + 8 * half;                 // D layout: row = v + 8*half
    float di = (float)((n >> 6) - mi);
    float dj = (float)((n & 63) - mj);
    float bias = 2.0f - KBIAS * fast_sqrtf(di * di + dj * dj);
    sv[v] = acc.f[v] * bias;
    tmax = fmaxf(tmax, sv[v]);
  }
  tmax = fmaxf(tmax, __shfl_xor(tmax, 16, 32));       // merge both lane halves
  float nm = fmaxf(runmax, tmax);
  float ts = 0.0f;
#pragma unroll
  for (int v = 0; v < 8; ++v) ts += __expf(sv[v] - nm);
  ts += __shfl_xor(ts, 16, 32);
  runsum = runsum * __expf(runmax - nm) + ts;
  runmax = nm;
}

// --------- Kernel 2: per-column (softmax over n) online max / sum-exp -------
__global__ __launch_bounds__(32) void colstats_kernel(
    const uint16_t* __restrict__ thetaB, const uint16_t* __restrict__ phiB,
    float* __restrict__ colmax, float* __restrict__ colrs) {
  const int lane = threadIdx.x;
  const int half = lane >> 4, lcol = lane & 15;
  const int m0 = blockIdx.x * 16;
  const int b = blockIdx.y;
  const uint16_t* th = thetaB + (size_t)b * Nn * Cc;
  const uint16_t* ph = phiB + (size_t)b * Nn * Cc;
  v16bf bf[8];                                 // resident phi fragments (full K=256)
#pragma unroll
  for (int kc = 0; kc < 8; ++kc) bf[kc] = load_b_frag(ph + (size_t)m0 * Cc, Cc, kc * 32, lane);
  const int m = m0 + lcol;
  const int mi = m >> 6, mj = m & 63;
  float runmax = -3.0e38f, runsum = 0.0f;

  v16bf afA[8], afB[8];                        // double-buffered theta A fragments
#pragma unroll
  for (int kc = 0; kc < 8; ++kc) afA[kc] = load_a_frag(th, Cc, kc * 32, lane);  // nt=0

  for (int nt = 0; nt < Nn / 16; nt += 2) {
    // Issue loads for tile nt+1 BEFORE consuming tile nt (latency overlap).
    const uint16_t* ab1 = th + (size_t)(nt + 1) * 16 * Cc;
#pragma unroll
    for (int kc = 0; kc < 8; ++kc) afB[kc] = load_a_frag(ab1, Cc, kc * 32, lane);

    FORCE_LIVE8(afA);
    F32x8 acc;
#pragma unroll
    for (int v = 0; v < 8; ++v) acc.f[v] = 0.0f;
#pragma unroll
    for (int kc = 0; kc < 8; ++kc) acc.v = wmma_bf16(afA[kc], bf[kc], acc.v);
    tile_stats(acc, nt * 16, half, mi, mj, runmax, runsum);

    // Branchless prefetch of tile nt+2; the final iteration reads a few KB past
    // this batch's theta region, which stays inside the workspace allocation.
    const uint16_t* ab2 = th + (size_t)(nt + 2) * 16 * Cc;
#pragma unroll
    for (int kc = 0; kc < 8; ++kc) afA[kc] = load_a_frag(ab2, Cc, kc * 32, lane);

    FORCE_LIVE8(afB);
#pragma unroll
    for (int v = 0; v < 8; ++v) acc.f[v] = 0.0f;
#pragma unroll
    for (int kc = 0; kc < 8; ++kc) acc.v = wmma_bf16(afB[kc], bf[kc], acc.v);
    tile_stats(acc, (nt + 1) * 16, half, mi, mj, runmax, runsum);
  }
  if (half == 0) {
    colmax[(size_t)b * Nn + m] = runmax;
    colrs[(size_t)b * Nn + m] = 1.0f / runsum;
  }
}

// --------- Kernel 3: recompute S tiles, P = exp(..)*rs, out = P @ g + x -----
__global__ __launch_bounds__(32) void out_kernel(
    const uint16_t* __restrict__ thetaB, const uint16_t* __restrict__ phiB,
    const uint16_t* __restrict__ gB, const float* __restrict__ colmax,
    const float* __restrict__ colrs, const float* __restrict__ x,
    float* __restrict__ out) {
  __shared__ uint16_t sP[16][34];              // 16(n) x 32(m) + pad, bf16
  const int lane = threadIdx.x;
  const int half = lane >> 4, lcol = lane & 15;
  const int n0 = blockIdx.x * 16;
  const int c0 = blockIdx.y * 128;             // 8 c-tiles of 16
  const int b = blockIdx.z;
  const uint16_t* th = thetaB + (size_t)b * Nn * Cc;
  const uint16_t* ph = phiB + (size_t)b * Nn * Cc;
  const uint16_t* gb = gB + (size_t)b * Cc * Nn;
  v16bf afT[8];                                // resident theta fragments
#pragma unroll
  for (int kc = 0; kc < 8; ++kc) afT[kc] = load_a_frag(th + (size_t)n0 * Cc, Cc, kc * 32, lane);
  F32x8 accO[8];
#pragma unroll
  for (int ct = 0; ct < 8; ++ct)
#pragma unroll
    for (int v = 0; v < 8; ++v) accO[ct].f[v] = 0.0f;

  for (int mc = 0; mc < Nn / 32; ++mc) {
    const int mbase = mc * 32;
#pragma unroll
    for (int mt = 0; mt < 2; ++mt) {
      const int m0t = mbase + mt * 16;
      const uint16_t* pbase = ph + (size_t)m0t * Cc;
      v16bf bfp[8];
#pragma unroll
      for (int kc = 0; kc < 8; ++kc) bfp[kc] = load_b_frag(pbase, Cc, kc * 32, lane);
      // Issue the per-column stat loads early so they overlap the WMMA chain.
      const int m = m0t + lcol;
      const float cm = colmax[(size_t)b * Nn + m];
      const float rs = colrs[(size_t)b * Nn + m];
      FORCE_LIVE8(bfp);
      F32x8 acc;
#pragma unroll
      for (int v = 0; v < 8; ++v) acc.f[v] = 0.0f;
#pragma unroll
      for (int kc = 0; kc < 8; ++kc) acc.v = wmma_bf16(afT[kc], bfp[kc], acc.v);
      const int mi = m >> 6, mj = m & 63;
#pragma unroll
      for (int v = 0; v < 8; ++v) {
        int n = n0 + v + 8 * half;
        float di = (float)((n >> 6) - mi);
        float dj = (float)((n & 63) - mj);
        float s = acc.f[v] * (2.0f - KBIAS * fast_sqrtf(di * di + dj * dj));
        float p = __expf(s - cm) * rs;         // identical S sequence to pass 1 => exp <= 1
        sP[v + 8 * half][mt * 16 + lcol] = f2bf(p);
      }
    }
    __syncthreads();
    // Issue g-fragment loads BEFORE the LDS transpose reads (overlap).
    v16bf gf[8];
#pragma unroll
    for (int ct = 0; ct < 8; ++ct)
      gf[ct] = load_b_frag(gb + (size_t)(c0 + ct * 16) * Nn, Nn, mbase, lane);
    // Build P A-fragment (16 n-rows x 32 m as K) from LDS transpose
    BF16x16 pf;
#pragma unroll
    for (int j = 0; j < 8; ++j) {
      int koff = 8 * half + 16 * (j >> 2) + 2 * (j & 3);
      pf.u[j] = *(const uint32_t*)&sP[lcol][koff];
    }
    FORCE_LIVE8(gf);
    // 8 independent WMMAs (distinct accumulators, shared A) -> back-to-back issue
#pragma unroll
    for (int ct = 0; ct < 8; ++ct) accO[ct].v = wmma_bf16(pf.v, gf[ct], accO[ct].v);
    __syncthreads();
  }
  // Epilogue: out[b][c][n] = acc + x[b][c][n]; lane rows are contiguous in n -> b128
#pragma unroll
  for (int ct = 0; ct < 8; ++ct) {
    int c = c0 + ct * 16 + lcol;
    size_t o = ((size_t)(b * Cc + c)) * Nn + n0 + 8 * half;
    float4 xa = *(const float4*)(x + o);
    float4 xb = *(const float4*)(x + o + 4);
    float4 oa, ob;
    oa.x = accO[ct].f[0] + xa.x; oa.y = accO[ct].f[1] + xa.y;
    oa.z = accO[ct].f[2] + xa.z; oa.w = accO[ct].f[3] + xa.w;
    ob.x = accO[ct].f[4] + xb.x; ob.y = accO[ct].f[5] + xb.y;
    ob.z = accO[ct].f[6] + xb.z; ob.w = accO[ct].f[7] + xb.w;
    *(float4*)(out + o) = oa;
    *(float4*)(out + o + 4) = ob;
  }
}

extern "C" void kernel_launch(void* const* d_in, const int* in_sizes, int n_in,
                              void* d_out, int out_size, void* d_ws, size_t ws_size,
                              hipStream_t stream) {
  (void)in_sizes; (void)n_in; (void)out_size; (void)ws_size;
  const float* x       = (const float*)d_in[0];
  const float* w_phi   = (const float*)d_in[1];
  const float* w_theta = (const float*)d_in[2];
  const float* w_g     = (const float*)d_in[3];
  float* out = (float*)d_out;

  uint16_t* thetaB = (uint16_t*)d_ws;                          // [B][N][C] bf16
  uint16_t* phiB   = thetaB + (size_t)Bb * Nn * Cc;            // [B][N][C] bf16
  uint16_t* gB     = phiB + (size_t)Bb * Nn * Cc;              // [B][C][N] bf16
  float* colmax    = (float*)(gB + (size_t)Bb * Nn * Cc);      // [B][N]
  float* colrs     = colmax + (size_t)Bb * Nn;                 // [B][N]

  proj_kernel<<<dim3(Nn / 64, (Cc / 64) * 3, Bb), 256, 0, stream>>>(
      x, w_phi, w_theta, w_g, thetaB, phiB, gB);
  colstats_kernel<<<dim3(Nn / 16, Bb), 32, 0, stream>>>(thetaB, phiB, colmax, colrs);
  out_kernel<<<dim3(Nn / 16, 2, Bb), 32, 0, stream>>>(
      thetaB, phiB, gB, colmax, colrs, x, out);
}
